// MultiSE3Transformer_6966436954873
// MI455X (gfx1250) — matrix operand order repaired and checked by reading, not compiled, not run on hardware
//
#include <hip/hip_runtime.h>
#include <hip/hip_bf16.h>

// MI455X (gfx1250) fused SE(3)-transformer.
// Design rationale (compile-only, reasoned from specs):
//  - Radial-MLP GEMMs ([E,16]@[16,64] -> silu -> @[64,288|576]) are ~30 GFLOP total:
//    run them on v_wmma_f32_16x16x32_f16 (f16 operands, f32 accum), one wave = one
//    16-edge M-tile, and consume the resulting per-edge weights directly from LDS so
//    the ~450 MB/layer of per-edge weights never touches HBM (HBM roundtrip would cost
//    ~80us at 23.3 TB/s vs ~10us of unavoidable gather/scatter traffic).
//  - Weights are pre-swizzled once per layer into WMMA B-fragment order (32 contiguous
//    bytes per lane per 16x16x32 tile) => single coalesced load per fragment, L2-resident.
//  - Softmax over dst segments: 3 passes (atomic fmax via int bit-trick, atomic add of
//    exp, then weighted scatter-add), all deterministic-per-call w.r.t. values.

typedef __attribute__((ext_vector_type(16))) _Float16 v16h;
typedef __attribute__((ext_vector_type(8)))  float    v8f;

#define SQ3F     1.7320508075688772f
#define INV_SQ3F 0.5773502691896258f

__device__ __forceinline__ v8f wmma_f16(v16h a, v16h b, v8f c) {
  return __builtin_amdgcn_wmma_f32_16x16x32_f16(false, a, false, b, (short)0, c, false, false);
}

// A fragment (16x32 f16) from a row-major half buffer [16][stride], starting at kbase.
// Layout per ISA 7.12.2: lanes 0-15 rows M=0..15 with K=kbase+0..7 (vgpr0-3) and
// kbase+16..23 (vgpr4-7); lanes 16-31 same rows, K offset +8.
__device__ __forceinline__ v16h load_afrag(const _Float16* buf, int stride, int kbase, int lane) {
  const int row = lane & 15;
  const _Float16* p = buf + row * stride + kbase + ((lane >> 4) << 3);
  v16h a;
#pragma unroll
  for (int j = 0; j < 8; ++j) { a[j] = p[j]; a[j + 8] = p[16 + j]; }
  return a;
}

__device__ __forceinline__ void atomicMaxFloat(float* addr, float v) {
  if (v >= 0.0f) atomicMax((int*)addr, __float_as_int(v));
  else           atomicMin((unsigned int*)addr, __float_as_uint(v));
}

// ---------------------------------------------------------------------------
// Generic e3nn o3.Linear: in [n, m0+3*m1] -> out [n, mo0+3*mo1]
// ---------------------------------------------------------------------------
__global__ void lin_kernel(const float* __restrict__ in, const float* __restrict__ W0,
                           const float* __restrict__ W1, float* __restrict__ out,
                           int n, int m0, int m1, int mo0, int mo1) {
  int i = blockIdx.x * blockDim.x + threadIdx.x;
  if (i >= n) return;
  const float* s = in + (size_t)i * (m0 + 3 * m1);
  const float* v = s + m0;
  float* o = out + (size_t)i * (mo0 + 3 * mo1);
  const float ns = rsqrtf((float)m0), nv = rsqrtf((float)m1);
  for (int w = 0; w < mo0; ++w) {
    float a = 0.f;
    for (int u = 0; u < m0; ++u) a += s[u] * W0[u * mo0 + w];
    o[w] = a * ns;
  }
  for (int w = 0; w < mo1; ++w)
    for (int k = 0; k < 3; ++k) {
      float a = 0.f;
      for (int u = 0; u < m1; ++u) a += v[u * 3 + k] * W1[u * mo1 + w];
      o[mo0 + w * 3 + k] = a * nv;
    }
}

// ---------------------------------------------------------------------------
// Pre-swizzle a [Ktot, Ncols] f32 weight matrix into f16 WMMA B-fragments.
// Tile (n, t): 32 lanes x 16 halves, element j of lane l = W[t*32+(j>>3)*16+(l>>4)*8+(j&7)][n*16+(l&15)].
// Rows >= Ktot are zero (K padding for the 16->32 first GEMM).
// ---------------------------------------------------------------------------
__global__ void pack_kernel(const float* __restrict__ W, _Float16* __restrict__ out,
                            int Ktot, int Ncols, int ksteps) {
  int idx = blockIdx.x * blockDim.x + threadIdx.x;
  int total = ksteps * (Ncols / 16) * 512;
  if (idx >= total) return;
  int j = idx & 15;
  int l = (idx >> 4) & 31;
  int b = idx >> 9;
  int t = b % ksteps;
  int n = b / ksteps;
  int k = t * 32 + ((j >> 3) << 4) + ((l >> 4) << 3) + (j & 7);
  int col = n * 16 + (l & 15);
  out[idx] = (k < Ktot) ? (_Float16)W[(size_t)k * Ncols + col] : (_Float16)0.0f;
}

__global__ void init_kernel(float* __restrict__ mx, float* __restrict__ z,
                            float* __restrict__ acc, int N) {
  int i = blockIdx.x * blockDim.x + threadIdx.x;
  if (i >= N) return;
  mx[i] = -__builtin_inff();
  z[i] = 0.f;
  for (int c = 0; c < 40; ++c) acc[(size_t)i * 40 + c] = 0.f;
}

__global__ void vec_add(float* __restrict__ x, const float* __restrict__ a, int n) {
  int i = blockIdx.x * blockDim.x + threadIdx.x;
  if (i < n) x[i] += a[i];
}

// ---------------------------------------------------------------------------
// Main fused edge kernel: one wave per 16 edges.
//   emb -> WMMA GEMM1 (x2 paths) -> silu -> WMMA GEMM2 (18+36 N-tiles) -> LDS
//   -> per-edge FullyConnectedTensorProduct (keys + values) -> logits, val.
// ---------------------------------------------------------------------------
__global__ __launch_bounds__(32) void edge_main(
    const float* __restrict__ x, const float* __restrict__ qn,
    const float* __restrict__ pos,
    const int* __restrict__ src, const int* __restrict__ dst,
    const _Float16* __restrict__ pk1, const _Float16* __restrict__ pk2,
    const _Float16* __restrict__ pv1, const _Float16* __restrict__ pv2,
    const float* __restrict__ dot0, const float* __restrict__ dot1,
    float* __restrict__ logits, float* __restrict__ cutB,
    float* __restrict__ val, float* __restrict__ mx, int E) {
  __shared__ _Float16 sA[16 * 32];    // emb, K-padded to 32
  __shared__ _Float16 sHk[16 * 64];   // silu hidden, key path
  __shared__ _Float16 sHv[16 * 64];   // silu hidden, value path
  __shared__ _Float16 sWk[16 * 288];  // per-edge key TP weights
  __shared__ _Float16 sWv[16 * 576];  // per-edge value TP weights
  __shared__ float sF[16 * 40];       // x[src]
  __shared__ float sQ[16 * 24];       // q[dst]
  __shared__ float sY[16 * 3];        // y1
  __shared__ float sVy[16 * 8];
  __shared__ float sKs[16 * 8];
  __shared__ float sKv[16 * 12];
  __shared__ float sVal[16 * 40];
  __shared__ float sCut[16];

  const int lane = threadIdx.x;
  const int el = lane & 15;     // edge within tile
  const int hw = lane >> 4;     // half-wave id (2 lanes per edge)
  int e = blockIdx.x * 16 + el;
  if (e >= E) e = E - 1;
  const int is = src[e], id = dst[e];

  // --- geometry / basis / cutoff ---
  float vx = pos[is * 3 + 0] - pos[id * 3 + 0];
  float vy = pos[is * 3 + 1] - pos[id * 3 + 1];
  float vz = pos[is * 3 + 2] - pos[id * 3 + 2];
  float dd = sqrtf(vx * vx + vy * vy + vz * vz + 1e-24f);
  if (hw == 0) {
    float inv = SQ3F / dd;
    sY[el * 3 + 0] = vx * inv;
    sY[el * 3 + 1] = vy * inv;
    sY[el * 3 + 2] = vz * inv;
    float xa = 10.0f * (1.0f - dd * 0.5f);
    float xs = fmaxf(xa, 1e-9f);
    sCut[el] = (xa > 0.0f) ? __expf(-1.0f / xs) : 0.0f;
  }
  {
    const float step = 2.0f / 17.0f, istep = 17.0f / 2.0f;
#pragma unroll
    for (int j = 0; j < 8; ++j) {
      int i = hw * 8 + j;
      float diff = (dd - step * (float)(i + 1)) * istep;
      sA[el * 32 + i] = (_Float16)(__expf(-diff * diff) * (4.0f / 1.12f));
      sA[el * 32 + 16 + i] = (_Float16)0.0f;  // K padding 16..31
    }
  }
  for (int j = 0; j < 20; ++j) sF[el * 40 + hw * 20 + j] = x[(size_t)is * 40 + hw * 20 + j];
  for (int j = 0; j < 12; ++j) sQ[el * 24 + hw * 12 + j] = qn[(size_t)id * 24 + hw * 12 + j];
  __syncthreads();

  // --- GEMM1: emb @ fc1 (K padded 16->32), both paths, silu with NB^-1/2 = 0.25 ---
  const int mrow = hw << 3;
  {
    v16h a1 = load_afrag(sA, 32, 0, lane);
#pragma unroll
    for (int n = 0; n < 4; ++n) {
      v16h bk = *(const v16h*)(pk1 + (size_t)(n * 32 + lane) * 16);
      v8f ck = {};
      ck = wmma_f16(a1, bk, ck);
#pragma unroll
      for (int r = 0; r < 8; ++r) {
        float h = ck[r] * 0.25f;
        sHk[(mrow + r) * 64 + n * 16 + el] = (_Float16)(h / (1.0f + __expf(-h)));
      }
      v16h bv = *(const v16h*)(pv1 + (size_t)(n * 32 + lane) * 16);
      v8f cv = {};
      cv = wmma_f16(a1, bv, cv);
#pragma unroll
      for (int r = 0; r < 8; ++r) {
        float h = cv[r] * 0.25f;
        sHv[(mrow + r) * 64 + n * 16 + el] = (_Float16)(h / (1.0f + __expf(-h)));
      }
    }
  }
  __syncthreads();

  // --- GEMM2: h @ fc2 (K=64 -> 2 ksteps), HID^-1/2 = 0.125 applied on store ---
  {
    v16h ak0 = load_afrag(sHk, 64, 0, lane);
    v16h ak1 = load_afrag(sHk, 64, 32, lane);
    for (int n = 0; n < 18; ++n) {
      v8f c = {};
      v16h b0 = *(const v16h*)(pk2 + (size_t)((n * 2 + 0) * 32 + lane) * 16);
      c = wmma_f16(ak0, b0, c);
      v16h b1 = *(const v16h*)(pk2 + (size_t)((n * 2 + 1) * 32 + lane) * 16);
      c = wmma_f16(ak1, b1, c);
#pragma unroll
      for (int r = 0; r < 8; ++r)
        sWk[(mrow + r) * 288 + n * 16 + el] = (_Float16)(c[r] * 0.125f);
    }
    v16h av0 = load_afrag(sHv, 64, 0, lane);
    v16h av1 = load_afrag(sHv, 64, 32, lane);
    for (int n = 0; n < 36; ++n) {
      v8f c = {};
      v16h b0 = *(const v16h*)(pv2 + (size_t)((n * 2 + 0) * 32 + lane) * 16);
      c = wmma_f16(av0, b0, c);
      v16h b1 = *(const v16h*)(pv2 + (size_t)((n * 2 + 1) * 32 + lane) * 16);
      c = wmma_f16(av1, b1, c);
#pragma unroll
      for (int r = 0; r < 8; ++r)
        sWv[(mrow + r) * 576 + n * 16 + el] = (_Float16)(c[r] * 0.125f);
    }
  }
  __syncthreads();

  // --- per-edge FullyConnectedTensorProduct (keys K0=8,K1=4; values 16,8) ---
  const float* fs = &sF[el * 40];
  const float* fv = &sF[el * 40 + 16];
  const float* y1 = &sY[el * 3];
  const _Float16* wk = &sWk[el * 288];
  const _Float16* wv = &sWv[el * 576];
  const float sS = 0.20412414523193154f;  // (m0+m1)^-1/2, fan = 24
  const float sV = 0.35355339059327373f;  // sqrt(3) * 24^-1/2

  if (hw == 0) {
    for (int u = 0; u < 8; ++u)
      sVy[el * 8 + u] = (fv[u * 3] * y1[0] + fv[u * 3 + 1] * y1[1] + fv[u * 3 + 2] * y1[2]) * INV_SQ3F;
  }
  __syncthreads();

  if (hw == 0) {
    // ks_ : W1 @ 0 (u*8+w, 128), W4 @ 224 (u*8+w)
    for (int w = 0; w < 8; ++w) {
      float a = 0.f;
      for (int u = 0; u < 16; ++u) a += fs[u] * (float)wk[u * 8 + w];
      for (int u = 0; u < 8; ++u)  a += sVy[el * 8 + u] * (float)wk[224 + u * 8 + w];
      sKs[el * 8 + w] = a * sS;
    }
    // vs_ : W1 @ 0 (u*16+w, 256), W4 @ 448
    for (int w = 0; w < 16; ++w) {
      float a = 0.f;
      for (int u = 0; u < 16; ++u) a += fs[u] * (float)wv[u * 16 + w];
      for (int u = 0; u < 8; ++u)  a += sVy[el * 8 + u] * (float)wv[448 + u * 16 + w];
      sVal[el * 40 + w] = a * sS;
    }
  } else {
    // kv_ : W2 @ 128 (u*4+w, 64), W3 @ 192 (u*4+w, 32)
    for (int w = 0; w < 4; ++w) {
      float t2 = 0.f;
      for (int u = 0; u < 16; ++u) t2 += fs[u] * (float)wk[128 + u * 4 + w];
      for (int k = 0; k < 3; ++k) {
        float a = t2 * y1[k] * INV_SQ3F;
        for (int u = 0; u < 8; ++u) a += fv[u * 3 + k] * (float)wk[192 + u * 4 + w];
        sKv[el * 12 + w * 3 + k] = a * sV;
      }
    }
    // vv_ : W2 @ 256 (u*8+w, 128), W3 @ 384 (u*8+w, 64)
    for (int w = 0; w < 8; ++w) {
      float t2 = 0.f;
      for (int u = 0; u < 16; ++u) t2 += fs[u] * (float)wv[256 + u * 8 + w];
      for (int k = 0; k < 3; ++k) {
        float a = t2 * y1[k] * INV_SQ3F;
        for (int u = 0; u < 8; ++u) a += fv[u * 3 + k] * (float)wv[384 + u * 8 + w];
        sVal[el * 40 + 16 + w * 3 + k] = a * sV;
      }
    }
  }
  __syncthreads();

  // --- logits + segment max ---
  if (hw == 0) {
    const float* qs = &sQ[el * 24];
    const float* qv = &sQ[el * 24 + 8];
    float lg0 = 0.f;
    for (int u = 0; u < 8; ++u) {
      float a = qs[u];
      for (int v = 0; v < 8; ++v) lg0 += a * sKs[el * 8 + v] * dot0[u * 8 + v];
    }
    float lg1 = 0.f;
    for (int u = 0; u < 4; ++u)
      for (int v = 0; v < 4; ++v) {
        float dp = qv[u * 3] * sKv[el * 12 + v * 3] + qv[u * 3 + 1] * sKv[el * 12 + v * 3 + 1] +
                   qv[u * 3 + 2] * sKv[el * 12 + v * 3 + 2];
        lg1 += dp * dot1[u * 4 + v];
      }
    float lg = (lg0 + lg1 * INV_SQ3F) * 0.11180339887498949f;  // (K0^2+K1^2)^-1/2
    logits[e] = lg;
    cutB[e] = sCut[el];
    atomicMaxFloat(&mx[id], lg);
  }
  for (int j = 0; j < 20; ++j)
    val[(size_t)e * 40 + hw * 20 + j] = sVal[el * 40 + hw * 20 + j];
}

__global__ void edge_softmax(const float* __restrict__ logits, const float* __restrict__ cutB,
                             const int* __restrict__ dst, const float* __restrict__ mx,
                             float* __restrict__ z, float* __restrict__ exB, int E) {
  int e = blockIdx.x * blockDim.x + threadIdx.x;
  if (e >= E) return;
  int d = dst[e];
  float m = mx[d];
  if (!__builtin_isfinite(m)) m = 0.0f;
  float ex = cutB[e] * __expf(logits[e] - m);
  exB[e] = ex;
  atomicAdd(&z[d], ex);
}

__global__ void edge_scatter(const float* __restrict__ exB, const float* __restrict__ val,
                             const int* __restrict__ dst, const float* __restrict__ z,
                             float* __restrict__ acc, int E) {
  int e = blockIdx.x * blockDim.x + threadIdx.x;
  if (e >= E) return;
  int d = dst[e];
  float zz = z[d];
  if (zz == 0.0f) zz = 1.0f;
  float alpha = exB[e] / zz;
  float w = sqrtf(fmaxf(alpha, 0.0f) + 1e-12f);
  const float* v = val + (size_t)e * 40;
  float* o = acc + (size_t)d * 40;
  for (int c = 0; c < 40; ++c) atomicAdd(&o[c], w * v[c]);
}

// ---------------------------------------------------------------------------
extern "C" void kernel_launch(void* const* d_in, const int* in_sizes, int n_in,
                              void* d_out, int out_size, void* d_ws, size_t ws_size,
                              hipStream_t stream) {
  const float* f    = (const float*)d_in[0];
  const float* pos  = (const float*)d_in[1];
  const float* lin0 = (const float*)d_in[2];
  const float* lin1 = (const float*)d_in[3];
  const float* hq0  = (const float*)d_in[4];
  const float* hq1  = (const float*)d_in[5];
  const float* fck1 = (const float*)d_in[6];
  const float* fck2 = (const float*)d_in[7];
  const float* fcv1 = (const float*)d_in[8];
  const float* fcv2 = (const float*)d_in[9];
  const float* dot0 = (const float*)d_in[10];
  const float* dot1 = (const float*)d_in[11];
  const float* rb0  = (const float*)d_in[12];
  const float* rb1  = (const float*)d_in[13];
  const float* re0  = (const float*)d_in[14];
  const float* re1  = (const float*)d_in[15];
  const int* esrc   = (const int*)d_in[16];
  const int* edst   = (const int*)d_in[17];
  const int N = in_sizes[0] / 40;
  const int E = in_sizes[16];

  // workspace layout (f32 units)
  float* ws  = (float*)d_ws;
  float* x   = ws;                          // N*40
  float* acc = x + (size_t)N * 40;          // N*40
  float* qb  = acc + (size_t)N * 40;        // N*24
  float* mx  = qb + (size_t)N * 24;         // N
  float* z   = mx + N;                      // N
  float* lg  = z + N;                       // E
  float* cut = lg + E;                      // E
  float* exb = cut + E;                     // E
  float* val = exb + E;                     // E*40
  _Float16* pk1 = (_Float16*)(val + (size_t)E * 40);  // 2048
  _Float16* pk2 = pk1 + 2048;                         // 18432
  _Float16* pv1 = pk2 + 18432;                        // 2048
  _Float16* pv2 = pv1 + 2048;                         // 36864

  dim3 b256(256);
  lin_kernel<<<dim3((N + 255) / 256), b256, 0, stream>>>(f, lin0, lin1, x, N, 16, 8, 16, 8);

  for (int l = 0; l < 2; ++l) {
    pack_kernel<<<dim3((2048 + 255) / 256), b256, 0, stream>>>(fck1 + (size_t)l * 16 * 64, pk1, 16, 64, 1);
    pack_kernel<<<dim3((18432 + 255) / 256), b256, 0, stream>>>(fck2 + (size_t)l * 64 * 288, pk2, 64, 288, 2);
    pack_kernel<<<dim3((2048 + 255) / 256), b256, 0, stream>>>(fcv1 + (size_t)l * 16 * 64, pv1, 16, 64, 1);
    pack_kernel<<<dim3((36864 + 255) / 256), b256, 0, stream>>>(fcv2 + (size_t)l * 64 * 576, pv2, 64, 576, 2);
    lin_kernel<<<dim3((N + 255) / 256), b256, 0, stream>>>(x, hq0 + (size_t)l * 16 * 8,
                                                           hq1 + (size_t)l * 8 * 4, qb, N, 16, 8, 8, 4);
    init_kernel<<<dim3((N + 255) / 256), b256, 0, stream>>>(mx, z, acc, N);
    edge_main<<<dim3((E + 15) / 16), dim3(32), 0, stream>>>(
        x, qb, pos, esrc, edst, pk1, pk2, pv1, pv2,
        dot0 + (size_t)l * 64, dot1 + (size_t)l * 16, lg, cut, val, mx, E);
    edge_softmax<<<dim3((E + 255) / 256), b256, 0, stream>>>(lg, cut, edst, mx, z, exb, E);
    edge_scatter<<<dim3((E + 255) / 256), b256, 0, stream>>>(exb, val, edst, z, acc, E);
    vec_add<<<dim3((N * 40 + 255) / 256), b256, 0, stream>>>(x, acc, N * 40);
  }

  float* out = (float*)d_out;
  lin_kernel<<<dim3((N + 255) / 256), b256, 0, stream>>>(x, rb0, rb1, out, N, 16, 8, 16, 8);
  lin_kernel<<<dim3((N + 255) / 256), b256, 0, stream>>>(x, re0, re1, out + (size_t)N * 40, N, 16, 8, 16, 8);
}
